// Decoder_54571854463116
// MI455X (gfx1250) — compile-verified, matrix operand order checked by reading, not run on hardware
//
#include <hip/hip_runtime.h>
#include <math.h>

typedef __attribute__((ext_vector_type(16))) _Float16 v16h;
typedef __attribute__((ext_vector_type(8)))  float    v8f;

// ---------------------------------------------------------------------------
// WMMA helpers (wave32, V_WMMA_F32_16X16X32_F16)
// A tile: 16 rows x 32 K, row-major source, ld = row stride (elements).
// B tile: weights stored pre-transposed as N x K row-major, so the load
// pattern is identical to A (per CDNA5 ISA 16-bit A-matrix layout).
// Codegen-verified: lowers to 2x global_load_b128 per tile per lane.
// ---------------------------------------------------------------------------
__device__ inline v16h load_tile(const _Float16* __restrict__ base, int ld) {
    const int lane = threadIdx.x & 31;
    const int r    = lane & 15;
    const int koff = (lane & 16) ? 8 : 0;
    const _Float16* p = base + (size_t)r * ld + koff;
    v16h v;
#pragma unroll
    for (int i = 0; i < 4; ++i) {      // VGPR i : K = koff+2i, koff+2i+1
        v[2*i]   = p[2*i];
        v[2*i+1] = p[2*i+1];
    }
#pragma unroll
    for (int i = 0; i < 4; ++i) {      // VGPR 4+i : K = 16+koff+2i
        v[8+2*i] = p[16+2*i];
        v[9+2*i] = p[16+2*i+1];
    }
    return v;
}

__device__ inline v8f wmma32(v16h a, v16h b, v8f c) {
    return __builtin_amdgcn_wmma_f32_16x16x32_f16(false, a, false, b,
                                                  (short)0, c, false, false);
}

// ---------------------------------------------------------------------------
// Weight transpose + f16 convert: dst[n*padK + k] = src[k*N + n] (0 if k>=K)
// ---------------------------------------------------------------------------
__global__ void transpose_h(const float* __restrict__ src, _Float16* __restrict__ dst,
                            int K, int N, int padK) {
    int idx = blockIdx.x * blockDim.x + threadIdx.x;
    int total = N * padK;
    if (idx >= total) return;
    int n = idx / padK, k = idx - n * padK;
    dst[idx] = (k < K) ? (_Float16)src[(size_t)k * N + n] : (_Float16)0.f;
}

// ---------------------------------------------------------------------------
// Embedding: x = latents + [centers, sin(p), cos(p)] @ embed_W + embed_b
// one block (192 threads) per token row
// ---------------------------------------------------------------------------
__global__ __launch_bounds__(192) void embed_add(
    const float* __restrict__ latents, const float* __restrict__ centers,
    const float* __restrict__ eW, const float* __restrict__ eB,
    float* __restrict__ X) {
    __shared__ float fin[51];
    int row = blockIdx.x;     // b*512 + m
    int t   = threadIdx.x;
    const float* c = centers + (size_t)row * 3;
    if (t < 3) {
        fin[t] = c[t];
    } else if (t < 51) {
        int e = (t < 27) ? (t - 3) : (t - 27);
        float crd = (e < 8) ? c[0] : ((e < 16) ? c[1] : c[2]);
        float p = crd * 3.14159265358979323846f * (float)(1 << (e & 7));
        fin[t] = (t < 27) ? sinf(p) : cosf(p);
    }
    __syncthreads();
    float acc = eB[t];
    for (int k = 0; k < 51; ++k) acc += fin[k] * eW[k * 192 + t];
    X[(size_t)row * 192 + t] = latents[(size_t)row * 192 + t] + acc;
}

// ---------------------------------------------------------------------------
// LayerNorm (C=192) -> f16 row-major, one wave per row
// ---------------------------------------------------------------------------
__global__ void ln_f16(const float* __restrict__ X, const float* __restrict__ g,
                       const float* __restrict__ bt, _Float16* __restrict__ OutH,
                       int rows) {
    const int C = 192;
    int wave = (blockIdx.x * blockDim.x + threadIdx.x) >> 5;
    int lane = threadIdx.x & 31;
    if (wave >= rows) return;
    const float* xr = X + (size_t)wave * C;
    float s1 = 0.f, s2 = 0.f;
    for (int c = lane; c < C; c += 32) { float v = xr[c]; s1 += v; s2 += v * v; }
    for (int off = 16; off; off >>= 1) { s1 += __shfl_xor(s1, off); s2 += __shfl_xor(s2, off); }
    float mu = s1 / C, var = s2 / C - mu * mu;
    float rs = rsqrtf(var + 1e-6f);
    for (int c = lane; c < C; c += 32)
        OutH[(size_t)wave * C + c] = (_Float16)((xr[c] - mu) * rs * g[c] + bt[c]);
}

// Final LayerNorm -> f16 transposed latents: latT[b][c][m]  (B-matrix layout feed)
__global__ void ln_latT(const float* __restrict__ X, const float* __restrict__ g,
                        const float* __restrict__ bt, _Float16* __restrict__ latT) {
    const int C = 192;
    int wave = (blockIdx.x * blockDim.x + threadIdx.x) >> 5;
    int lane = threadIdx.x & 31;
    if (wave >= 2048) return;
    const float* xr = X + (size_t)wave * C;
    float s1 = 0.f, s2 = 0.f;
    for (int c = lane; c < C; c += 32) { float v = xr[c]; s1 += v; s2 += v * v; }
    for (int off = 16; off; off >>= 1) { s1 += __shfl_xor(s1, off); s2 += __shfl_xor(s2, off); }
    float mu = s1 / C, var = s2 / C - mu * mu;
    float rs = rsqrtf(var + 1e-6f);
    int b = wave >> 9, m = wave & 511;
    for (int c = lane; c < C; c += 32)
        latT[((size_t)b * 192 + c) * 512 + m] =
            (_Float16)((xr[c] - mu) * rs * g[c] + bt[c]);
}

// ---------------------------------------------------------------------------
// Generic WMMA GEMM: out = A(f16, MxK row-major) * BT(f16, NxK row-major)^T
// mode 0: outF = acc + bias            (fp32, ld=N)
// mode 1: resid += gamma * (acc+bias)  (fp32, ld=ldr)
// mode 2: outH  = gelu(acc + bias)     (f16, ld=N)
// one wave per 16x16 tile, 4 waves per block
// ---------------------------------------------------------------------------
__global__ __launch_bounds__(128) void gemm16(
    const _Float16* __restrict__ A, const _Float16* __restrict__ BT,
    const float* __restrict__ bias, int Mrows, int N, int K, int mode,
    float* __restrict__ outF, _Float16* __restrict__ outH,
    float* __restrict__ resid, int ldr, const float* __restrict__ gamma) {
    int ntiles = N >> 4;
    int tile = blockIdx.x * 4 + (threadIdx.x >> 5);
    if (tile >= (Mrows >> 4) * ntiles) return;
    int rt = tile / ntiles, ct = tile - rt * ntiles;
    const _Float16* Ab = A  + (size_t)rt * 16 * K;
    const _Float16* Bb = BT + (size_t)ct * 16 * K;
    v8f acc = {0.f,0.f,0.f,0.f,0.f,0.f,0.f,0.f};
    for (int k = 0; k < K; k += 32) {
        v16h a = load_tile(Ab + k, K);
        v16h b = load_tile(Bb + k, K);
        acc = wmma32(a, b, acc);
    }
    int lane = threadIdx.x & 31;
    int n = lane & 15, m0 = (lane & 16) ? 8 : 0;
    int col = ct * 16 + n;
    float bv = bias[col];
#pragma unroll
    for (int i = 0; i < 8; ++i) {
        int row = rt * 16 + m0 + i;
        float v = acc[i] + bv;
        if (mode == 0) {
            outF[(size_t)row * N + col] = v;
        } else if (mode == 1) {
            resid[(size_t)row * ldr + col] += gamma[col] * v;
        } else {
            outH[(size_t)row * N + col] =
                (_Float16)(0.5f * v * (1.0f + erff(v * 0.70710678118654752f)));
        }
    }
}

// ---------------------------------------------------------------------------
// Attention: one thread per (b,h,q), online softmax, head dim 32
// QKV fp32 row-major ld=576 (q:0, k:192, v:384); output f16 (b,m,h*32+d)
// ---------------------------------------------------------------------------
__global__ void attn(const float* __restrict__ QKV, _Float16* __restrict__ O16) {
    int tid = blockIdx.x * blockDim.x + threadIdx.x;
    if (tid >= 4 * 6 * 512) return;
    int q = tid & 511, bh = tid >> 9;
    int h = bh % 6, b = bh / 6;
    const float* base = QKV + (size_t)b * 512 * 576;
    float qv[32];
    const float* qr = base + (size_t)q * 576 + h * 32;
#pragma unroll
    for (int d = 0; d < 32; ++d) qv[d] = qr[d];
    float m = -3.4e38f, l = 0.f, o[32];
#pragma unroll
    for (int d = 0; d < 32; ++d) o[d] = 0.f;
    for (int k = 0; k < 512; ++k) {
        const float* kr = base + (size_t)k * 576 + 192 + h * 32;
        float s = 0.f;
#pragma unroll
        for (int d = 0; d < 32; ++d) s += qv[d] * kr[d];
        s *= 0.17677669529663687f;   // 32^-0.5
        float nm  = fmaxf(m, s);
        float esc = expf(m - nm);
        float p   = expf(s - nm);
        l = l * esc + p;
        const float* vr = kr + 192;
#pragma unroll
        for (int d = 0; d < 32; ++d) o[d] = o[d] * esc + p * vr[d];
        m = nm;
    }
    _Float16* orow = O16 + (size_t)(b * 512 + q) * 192 + h * 32;
    float li = 1.f / l;
#pragma unroll
    for (int d = 0; d < 32; ++d) orow[d] = (_Float16)(o[d] * li);
}

// ---------------------------------------------------------------------------
// Stage C: fused per-sample decoder. One 128-thread block (4 waves) handles
// 32 samples of batch b, cooperating on shared 64KB LDS:
//   phase 1: softmax weights -> bufA (32x512 f16), 1/sum -> invp (in bufB hole)
//   phase 2: h0 static cols (sample, sin, cos, zero-pad) -> bufB (ld 256)
//   phase 3: lat_s = (w*inv) @ latT  (WMMA) -> h0 cols 51..242
//   phase 4: h1 = relu(h0 @ W1 + b1) (WMMA, K=256 padded) -> bufA (ld 512)
//   phase 5: h2 = relu(h1 @ W2 + b2) (WMMA, K=512)        -> bufB (ld 512)
//   phase 6: preds = h2 . w3 + b3
// WMMA tile loops are strided across the 4 waves; rows split 8/wave elsewhere.
// 64KB LDS/block, 4 waves/block -> ~20 waves per WGP for latency hiding.
// ---------------------------------------------------------------------------
__global__ __launch_bounds__(128) void stageC(
    const float* __restrict__ samples, const float* __restrict__ centers,
    const _Float16* __restrict__ latT, const _Float16* __restrict__ W1T,
    const float* __restrict__ b1, const _Float16* __restrict__ W2T,
    const float* __restrict__ b2, const float* __restrict__ w3,
    const float* __restrict__ b3, const float* __restrict__ log_sigma,
    float* __restrict__ out) {
    __shared__ _Float16 bufA[32 * 512];
    __shared__ _Float16 bufB[32 * 512];
    // rows*256..rows*512 of bufB are unused until phase 5: stash 1/sum there
    float* invp = (float*)(bufB + 32 * 256);

    const int lane = threadIdx.x & 31;
    const int wv   = threadIdx.x >> 5;        // wave 0..3
    const int b    = blockIdx.x >> 10;        // 1024 tiles of 32 per batch
    const int s0   = (blockIdx.x & 1023) * 32;
    const float sigma = expf(log_sigma[0]);

    // register-held centers: lane owns c where c%32==lane (each wave needs all)
    float cx[16], cy[16], cz[16];
    const float* cb = centers + (size_t)b * 512 * 3;
#pragma unroll
    for (int j = 0; j < 16; ++j) {
        int c = j * 32 + lane;
        cx[j] = cb[c * 3]; cy[j] = cb[c * 3 + 1]; cz[j] = cb[c * 3 + 2];
    }
    const float* sb = samples + (size_t)b * 32768 * 3;

    // phase 1: softmax weights for rows wv*8 .. wv*8+7
    for (int r0 = 0; r0 < 8; ++r0) {
        int r = wv * 8 + r0;
        float sx = sb[(s0 + r) * 3], sy = sb[(s0 + r) * 3 + 1], sz = sb[(s0 + r) * 3 + 2];
        float lv[16]; float mx = -3.4e38f;
#pragma unroll
        for (int j = 0; j < 16; ++j) {
            float dx = sx - cx[j], dy = sy - cy[j], dz = sz - cz[j];
            lv[j] = -(dx * dx + dy * dy + dz * dz) * sigma;
            mx = fmaxf(mx, lv[j]);
        }
        for (int off = 16; off; off >>= 1) mx = fmaxf(mx, __shfl_xor(mx, off));
        float s = 0.f;
#pragma unroll
        for (int j = 0; j < 16; ++j) {
            float e = expf(lv[j] - mx);
            s += e;
            bufA[r * 512 + j * 32 + lane] = (_Float16)e;
        }
        for (int off = 16; off; off >>= 1) s += __shfl_xor(s, off);
        if (lane == 0) invp[r] = 1.f / s;
    }

    // phase 2: h0 static columns (sample coords + fourier, zero pad)
    for (int r0 = 0; r0 < 8; ++r0) {
        int r = wv * 8 + r0;
        float sx = sb[(s0 + r) * 3], sy = sb[(s0 + r) * 3 + 1], sz = sb[(s0 + r) * 3 + 2];
#pragma unroll
        for (int j = 0; j < 8; ++j) {
            int col = lane + 32 * j;
            if (col >= 51 && col < 243) continue;   // filled by lat_s epilogue
            float val;
            if (col < 3) {
                val = (col == 0) ? sx : ((col == 1) ? sy : sz);
            } else if (col >= 243) {
                val = 0.f;
            } else {
                int e = (col < 27) ? col - 3 : col - 27;
                float crd = (e < 8) ? sx : ((e < 16) ? sy : sz);
                float p = crd * 3.14159265358979323846f * (float)(1 << (e & 7));
                val = (col < 27) ? sinf(p) : cosf(p);
            }
            bufB[r * 256 + col] = (_Float16)val;
        }
    }
    __syncthreads();

    const int n  = lane & 15;
    const int m0 = (lane & 16) ? 8 : 0;

    // phase 3: lat_s = w @ lat (scale by 1/sum in epilogue) -> h0 cols 51..242
    // 24 tiles (2 row-tiles x 12 col-tiles) strided across 4 waves
    const _Float16* lb = latT + (size_t)b * 192 * 512;
    for (int tile = wv; tile < 24; tile += 4) {
        int t = tile / 12, nt = tile - t * 12;
        v8f acc = {0.f,0.f,0.f,0.f,0.f,0.f,0.f,0.f};
        const _Float16* Bb = lb + (size_t)nt * 16 * 512;
        for (int k = 0; k < 512; k += 32) {
            v16h a  = load_tile(&bufA[t * 16 * 512 + k], 512);
            v16h bm = load_tile(Bb + k, 512);
            acc = wmma32(a, bm, acc);
        }
#pragma unroll
        for (int i = 0; i < 8; ++i) {
            int row = t * 16 + m0 + i;
            bufB[row * 256 + 51 + nt * 16 + n] = (_Float16)(acc[i] * invp[row]);
        }
    }
    __syncthreads();

    // phase 4: h1 = relu(h0 @ W1 + b1), K padded to 256 -> bufA (ld 512)
    for (int tile = wv; tile < 64; tile += 4) {
        int t = tile >> 5, nt = tile & 31;
        v8f acc = {0.f,0.f,0.f,0.f,0.f,0.f,0.f,0.f};
        const _Float16* Bb = W1T + (size_t)nt * 16 * 256;
        for (int k = 0; k < 256; k += 32) {
            v16h a  = load_tile(&bufB[t * 16 * 256 + k], 256);
            v16h bm = load_tile(Bb + k, 256);
            acc = wmma32(a, bm, acc);
        }
        int col = nt * 16 + n; float bv = b1[col];
#pragma unroll
        for (int i = 0; i < 8; ++i) {
            int row = t * 16 + m0 + i;
            bufA[row * 512 + col] = (_Float16)fmaxf(acc[i] + bv, 0.f);
        }
    }
    __syncthreads();

    // phase 5: h2 = relu(h1 @ W2 + b2), K=512 -> bufB (ld 512)
    for (int tile = wv; tile < 64; tile += 4) {
        int t = tile >> 5, nt = tile & 31;
        v8f acc = {0.f,0.f,0.f,0.f,0.f,0.f,0.f,0.f};
        const _Float16* Bb = W2T + (size_t)nt * 16 * 512;
        for (int k = 0; k < 512; k += 32) {
            v16h a  = load_tile(&bufA[t * 16 * 512 + k], 512);
            v16h bm = load_tile(Bb + k, 512);
            acc = wmma32(a, bm, acc);
        }
        int col = nt * 16 + n; float bv = b2[col];
#pragma unroll
        for (int i = 0; i < 8; ++i) {
            int row = t * 16 + m0 + i;
            bufB[row * 512 + col] = (_Float16)fmaxf(acc[i] + bv, 0.f);
        }
    }
    __syncthreads();

    // phase 6: preds = h2 . w3 + b3, rows split 8 per wave
    float b3v = b3[0];
    for (int r0 = 0; r0 < 8; ++r0) {
        int r = wv * 8 + r0;
        float s = 0.f;
#pragma unroll
        for (int j = 0; j < 16; ++j) {
            int k = j * 32 + lane;
            s += (float)bufB[r * 512 + k] * w3[k];
        }
        for (int off = 16; off; off >>= 1) s += __shfl_xor(s, off);
        if (lane == 0) out[(size_t)b * 32768 + s0 + r] = s + b3v;
    }
    if (blockIdx.x == 0 && threadIdx.x == 0) out[131072] = sigma;  // 2nd output
}

// ---------------------------------------------------------------------------
extern "C" void kernel_launch(void* const* d_in, const int* in_sizes, int n_in,
                              void* d_out, int out_size, void* d_ws, size_t ws_size,
                              hipStream_t stream) {
    (void)in_sizes; (void)n_in; (void)out_size; (void)ws_size;
    const float* latents   = (const float*)d_in[0];
    const float* centers   = (const float*)d_in[1];
    const float* samples   = (const float*)d_in[2];
    const float* log_sigma = (const float*)d_in[3];
    const float* embed_W   = (const float*)d_in[4];
    const float* embed_b   = (const float*)d_in[5];
    const float* ln1_g     = (const float*)d_in[6];
    const float* ln1_b     = (const float*)d_in[7];
    const float* qkv_W     = (const float*)d_in[8];
    const float* qkv_b     = (const float*)d_in[9];
    const float* proj_W    = (const float*)d_in[10];
    const float* proj_b    = (const float*)d_in[11];
    const float* g1        = (const float*)d_in[12];
    const float* g2        = (const float*)d_in[13];
    const float* ln2_g     = (const float*)d_in[14];
    const float* ln2_b     = (const float*)d_in[15];
    const float* fc1_W     = (const float*)d_in[16];
    const float* fc1_b     = (const float*)d_in[17];
    const float* fc2_W     = (const float*)d_in[18];
    const float* fc2_b     = (const float*)d_in[19];
    const float* norm_g    = (const float*)d_in[20];
    const float* norm_b    = (const float*)d_in[21];
    const float* mlp1_W    = (const float*)d_in[22];
    const float* mlp1_b    = (const float*)d_in[23];
    const float* mlp2_W    = (const float*)d_in[24];
    const float* mlp2_b    = (const float*)d_in[25];
    const float* mlp3_W    = (const float*)d_in[26];
    const float* mlp3_b    = (const float*)d_in[27];

    char* ws = (char*)d_ws;
    size_t off = 0;
    auto alloc = [&](size_t bytes) -> void* {
        void* p = ws + off;
        off = (off + bytes + 255) & ~(size_t)255;
        return p;
    };
    float*     x      = (float*)    alloc((size_t)2048 * 192 * 4);
    float*     qkvbuf = (float*)    alloc((size_t)2048 * 576 * 4);
    _Float16*  h16a   = (_Float16*) alloc((size_t)2048 * 192 * 2);
    _Float16*  o16    = (_Float16*) alloc((size_t)2048 * 192 * 2);
    _Float16*  hbig   = (_Float16*) alloc((size_t)2048 * 768 * 2);
    _Float16*  latT   = (_Float16*) alloc((size_t)4 * 192 * 512 * 2);
    _Float16*  qkvT   = (_Float16*) alloc((size_t)6 * 576 * 192 * 2);
    _Float16*  projT  = (_Float16*) alloc((size_t)6 * 192 * 192 * 2);
    _Float16*  fc1T   = (_Float16*) alloc((size_t)6 * 768 * 192 * 2);
    _Float16*  fc2T   = (_Float16*) alloc((size_t)6 * 192 * 768 * 2);
    _Float16*  W1T    = (_Float16*) alloc((size_t)512 * 256 * 2);
    _Float16*  W2T    = (_Float16*) alloc((size_t)512 * 512 * 2);

    auto tlaunch = [&](const float* src, _Float16* dst, int K, int N, int padK) {
        int total = N * padK;
        transpose_h<<<(total + 255) / 256, 256, 0, stream>>>(src, dst, K, N, padK);
    };
    for (int i = 0; i < 6; ++i) {
        tlaunch(qkv_W  + (size_t)i * 192 * 576, qkvT + (size_t)i * 576 * 192, 192, 576, 192);
        tlaunch(proj_W + (size_t)i * 192 * 192, projT + (size_t)i * 192 * 192, 192, 192, 192);
        tlaunch(fc1_W  + (size_t)i * 192 * 768, fc1T + (size_t)i * 768 * 192, 192, 768, 192);
        tlaunch(fc2_W  + (size_t)i * 768 * 192, fc2T + (size_t)i * 192 * 768, 768, 192, 768);
    }
    tlaunch(mlp1_W, W1T, 243, 512, 256);
    tlaunch(mlp2_W, W2T, 512, 512, 512);

    embed_add<<<2048, 192, 0, stream>>>(latents, centers, embed_W, embed_b, x);

    for (int i = 0; i < 6; ++i) {
        ln_f16<<<256, 256, 0, stream>>>(x, ln1_g + i * 192, ln1_b + i * 192, h16a, 2048);
        gemm16<<<(128 * 36) / 4, 128, 0, stream>>>(h16a, qkvT + (size_t)i * 576 * 192,
            qkv_b + i * 576, 2048, 576, 192, 0, qkvbuf, nullptr, nullptr, 0, nullptr);
        attn<<<48, 256, 0, stream>>>(qkvbuf, o16);
        gemm16<<<(128 * 12) / 4, 128, 0, stream>>>(o16, projT + (size_t)i * 192 * 192,
            proj_b + i * 192, 2048, 192, 192, 1, nullptr, nullptr, x, 192, g1 + i * 192);
        ln_f16<<<256, 256, 0, stream>>>(x, ln2_g + i * 192, ln2_b + i * 192, h16a, 2048);
        gemm16<<<(128 * 48) / 4, 128, 0, stream>>>(h16a, fc1T + (size_t)i * 768 * 192,
            fc1_b + i * 768, 2048, 768, 192, 2, nullptr, hbig, nullptr, 0, nullptr);
        gemm16<<<(128 * 12) / 4, 128, 0, stream>>>(hbig, fc2T + (size_t)i * 192 * 768,
            fc2_b + i * 192, 2048, 192, 768, 1, nullptr, nullptr, x, 192, g2 + i * 192);
    }
    ln_latT<<<256, 256, 0, stream>>>(x, norm_g, norm_b, latT);

    stageC<<<4096, 128, 0, stream>>>(samples, centers, latT, W1T, mlp1_b,
                                     W2T, mlp2_b, mlp3_W, mlp3_b, log_sigma,
                                     (float*)d_out);
}
